// KNNLogitHead_10007273799761
// MI455X (gfx1250) — matrix operand order repaired and checked by reading, not compiled, not run on hardware
//
#include <hip/hip_runtime.h>

typedef __attribute__((ext_vector_type(16))) _Float16 v16h;
typedef __attribute__((ext_vector_type(8)))  float    v8f;

// Branch-free gelu: 0.5*x*(1+erf(x/sqrt(2))) with Abramowitz-Stegun 7.1.26
// erf approximation (|abs err| <= 1.5e-7). No branches, no exec-mask churn:
// v_rcp_f32 + 5 FMA Horner + v_exp_f32 + v_bfi (copysign).
__device__ __forceinline__ float gelu_fast(float x) {
    float z    = fabsf(x) * 0.70710678118654752f;          // |x|/sqrt(2)
    float t    = __builtin_amdgcn_rcpf(fmaf(0.3275911f, z, 1.0f));
    float poly = t * fmaf(t, fmaf(t, fmaf(t, fmaf(t, 1.061405429f,
                          -1.453152027f), 1.421413741f),
                          -0.284496736f), 0.254829592f);
    float e    = __expf(-z * z);                            // underflows to 0, no branch
    float erfv = fmaf(-poly, e, 1.0f);                      // erf(|x|/sqrt2)
    float se   = copysignf(erfv, x);
    return 0.5f * x * (1.0f + se);
}

// 8-way select by per-lane index c (0..7); 7 v_cndmask, keeps arrays in regs
__device__ __forceinline__ float sel8(int c, const float* v) {
    float a0 = (c & 1) ? v[1] : v[0];
    float a1 = (c & 1) ? v[3] : v[2];
    float a2 = (c & 1) ? v[5] : v[4];
    float a3 = (c & 1) ? v[7] : v[6];
    float b0 = (c & 2) ? a1 : a0;
    float b1 = (c & 2) ? a3 : a2;
    return (c & 4) ? b1 : b0;
}

__device__ __forceinline__ v8f wmma_f16(v16h a, v16h b, v8f c) {
    // D = A(16x32 f16) * B(32x16 f16) + C(16x16 f32)
    return __builtin_amdgcn_wmma_f32_16x16x32_f16(
        /*neg_a=*/false, a, /*neg_b=*/false, b,
        /*c_mod=*/(short)0, c, /*reuse_a=*/false, /*reuse_b=*/false);
}

__global__ void __launch_bounds__(256)
knn_logit_head_kernel(const float* __restrict__ vals,   // (B,16)
                      const float* __restrict__ clsf,   // (B,16)
                      const float* __restrict__ W1,     // (1,32)
                      const float* __restrict__ b1,     // (32)
                      const float* __restrict__ W2,     // (32,32)
                      const float* __restrict__ b2,     // (32)
                      const float* __restrict__ Wr1,    // (33,32)
                      const float* __restrict__ br1,    // (32)
                      const float* __restrict__ Wr2,    // (32,1)
                      const float* __restrict__ br2p,   // (1)
                      const float* __restrict__ rs1p,   // residual_scale
                      const float* __restrict__ rs2p,   // residual_scale2
                      float* __restrict__ out,          // 3 * B * 8
                      int B, int nWaves)
{
    // per-wave pooled buffer: 8 classes x 36 f32 (16B-aligned rows, spread banks)
    __shared__ float lds_pool[8][8 * 36];

    const int tid  = threadIdx.x;
    const int lane = tid & 31;
    const int wib  = tid >> 5;
    const int wid  = blockIdx.x * (blockDim.x >> 5) + wib;
    float* pool = lds_pool[wib];

    const int hi = (lane >= 16) ? 1 : 0;   // half-wave
    const int n0 = lane & 15;              // B/C/D column index
    const int kk = lane & 15;              // neighbor owned for loads/ballots

    // ---- per-wave invariant preloads (A-layout feature slots for this lane) ----
    // j(r,h) = 2*(r&3) + h + (r>=4 ? 16 : 0) + (hi ? 8 : 0)
    float w1v[16], b1v[16];
#pragma unroll
    for (int r = 0; r < 8; ++r) {
#pragma unroll
        for (int h = 0; h < 2; ++h) {
            int j = 2 * (r & 3) + h + ((r >= 4) ? 16 : 0) + (hi ? 8 : 0);
            w1v[2 * r + h] = W1[j];
            b1v[2 * r + h] = b1[j];
        }
    }
    // B-matrix layout (32x16 f16): lanes 0-15 hold K=0..7,16..23 (packed pairs in
    // regs 0..7), lanes 16-31 hold K=8..15,24..31; lane's column n = n0 + 16*tile
    v16h bW2[2], bR1[2];
#pragma unroll
    for (int t = 0; t < 2; ++t) {
        int n  = n0 + 16 * t;
#pragma unroll
        for (int v = 0; v < 8; ++v) {
            int kb = hi * 8 + ((v >= 4) ? 8 : 0);
            bW2[t][2 * v]     = (_Float16)W2[(kb + 2 * v) * 32 + n];
            bW2[t][2 * v + 1] = (_Float16)W2[(kb + 2 * v + 1) * 32 + n];
            bR1[t][2 * v]     = (_Float16)Wr1[(kb + 2 * v) * 32 + n];
            bR1[t][2 * v + 1] = (_Float16)Wr1[(kb + 2 * v + 1) * 32 + n];
        }
    }
    const float b2v0  = b2[n0],            b2v1  = b2[16 + n0];
    const float br1v0 = br1[n0],           br1v1 = br1[16 + n0];
    const float wr2v0 = Wr2[n0],           wr2v1 = Wr2[16 + n0];
    const float w32v0 = Wr1[32 * 32 + n0], w32v1 = Wr1[32 * 32 + 16 + n0];
    const float rs1 = rs1p[0], rs2 = rs2p[0], br2s = br2p[0];

    for (int s = wid; s < B; s += nWaves) {
        // ---- zero pooled[8][0..31] (256 dwords, 8/lane, b128 stores) ----
        {
            int c = lane >> 2, q = (lane & 3) * 8;
            float4 z = make_float4(0.f, 0.f, 0.f, 0.f);
            *(float4*)&pool[c * 36 + q]     = z;
            *(float4*)&pool[c * 36 + q + 4] = z;
        }

        const float v  = vals[(size_t)s * 16 + kk];
        const float cf = clsf[(size_t)s * 16 + kk];
        const int   ci = (int)cf;   // exact small integer 0..7

        // ---- counts + class_sum (exact f32, matches the degenerate softmax) ----
        // Lanes 16..31 duplicate lanes 0..15, so a 4-step xor reduce per
        // 16-lane half gives the full sum in every lane.
        float cnt[8], csum[8];
#pragma unroll
        for (int c = 0; c < 8; ++c) {
            unsigned long long m = __ballot(ci == c);
            cnt[c] = (float)__popc((unsigned)m & 0xFFFFu);
            float t = (ci == c) ? v : 0.0f;
            t += __shfl_xor(t, 1, 32);
            t += __shfl_xor(t, 2, 32);
            t += __shfl_xor(t, 4, 32);
            t += __shfl_xor(t, 8, 32);
            csum[c] = t;
        }

        // ---- stage A: h = gelu(v*W1+b1), built directly in WMMA A layout ----
        v16h a;
#pragma unroll
        for (int r = 0; r < 8; ++r) {
            float h0 = gelu_fast(fmaf(v, w1v[2 * r],     b1v[2 * r]));
            float h1 = gelu_fast(fmaf(v, w1v[2 * r + 1], b1v[2 * r + 1]));
            a[2 * r]     = (_Float16)h0;
            a[2 * r + 1] = (_Float16)h1;
        }

        // ---- phi GEMM: (16x32) @ W2(32x32) via 2 WMMAs ----
        v8f cz = {};
        v8f d0 = wmma_f16(a, bW2[0], cz);
        v8f d1 = wmma_f16(a, bW2[1], cz);

        // row (neighbor) class for this lane's D rows: m = i + 8*hi
        int rc[8];
#pragma unroll
        for (int i = 0; i < 8; ++i) rc[i] = __shfl(ci, i + hi * 8, 32);

        // ---- phi_out = gelu(D+b2); one-hot pooling via LDS f32 atomics ----
#pragma unroll
        for (int i = 0; i < 8; ++i) {
            float p0 = gelu_fast(d0[i] + b2v0);
            float p1 = gelu_fast(d1[i] + b2v1);
            int  c  = rc[i];
            atomicAdd(&pool[c * 36 + n0],      p0);
            atomicAdd(&pool[c * 36 + 16 + n0], p1);
        }
        // DS ops are in-order within a wave; fence the compiler + hw counter
        asm volatile("s_wait_dscnt 0" ::: "memory");

        // ---- rho A matrix (16x32 f16): rows = classes (8..15 dup/unused) ----
        const int   m2  = lane & 7;
        const int   kb2 = hi ? 8 : 0;
        const float* prow = &pool[m2 * 36];
        float4 pa = *(const float4*)&prow[kb2];
        float4 pb = *(const float4*)&prow[kb2 + 4];
        float4 pc = *(const float4*)&prow[kb2 + 16];
        float4 pd = *(const float4*)&prow[kb2 + 20];
        float pv[16] = {pa.x, pa.y, pa.z, pa.w, pb.x, pb.y, pb.z, pb.w,
                        pc.x, pc.y, pc.z, pc.w, pd.x, pd.y, pd.z, pd.w};
        v16h a2;
#pragma unroll
        for (int i = 0; i < 16; ++i) a2[i] = (_Float16)pv[i];

        // ---- rho GEMM: pooled(:, 0:32) @ Wr1(0:32, :) via 2 WMMAs ----
        v8f r0 = wmma_f16(a2, bR1[0], cz);
        v8f r1 = wmma_f16(a2, bR1[1], cz);
        // fold the counts row (K=32): += counts[c] * Wr1[32][n]
#pragma unroll
        for (int i = 0; i < 8; ++i) {
            r0[i] = fmaf(cnt[i], w32v0, r0[i]);
            r1[i] = fmaf(cnt[i], w32v1, r1[i]);
        }

        // ---- delta[c] = gelu(. + br1) @ Wr2 + br2 (reduce over 32 hidden) ----
        float delta[8];
#pragma unroll
        for (int i = 0; i < 8; ++i) {
            float g0 = gelu_fast(r0[i] + br1v0);
            float g1 = gelu_fast(r1[i] + br1v1);
            float t  = fmaf(g0, wr2v0, g1 * wr2v1);
            t += __shfl_xor(t, 1, 32);
            t += __shfl_xor(t, 2, 32);
            t += __shfl_xor(t, 4, 32);
            t += __shfl_xor(t, 8, 32);   // valid within lanes 0..15
            delta[i] = t + br2s;
        }

        // ---- outputs: [logits | class_sum | counts], each (B,8) ----
        if (lane < 24) {
            int arr = lane >> 3, c = lane & 7;
            float dsel = sel8(c, delta);  // used only by lanes 0..7 (valid half)
            float ssel = sel8(c, csum);
            float nsel = sel8(c, cnt);
            float val = (arr == 0) ? fmaf(rs1, dsel, ssel * rs2)
                                   : ((arr == 1) ? ssel : nsel);
            out[(size_t)arr * (size_t)B * 8 + (size_t)s * 8 + c] = val;
        }
    }
}

extern "C" void kernel_launch(void* const* d_in, const int* in_sizes, int n_in,
                              void* d_out, int out_size, void* d_ws, size_t ws_size,
                              hipStream_t stream) {
    (void)n_in; (void)out_size; (void)d_ws; (void)ws_size;
    const int B = in_sizes[0] / 16;

    const float* vals = (const float*)d_in[0];
    const float* clsf = (const float*)d_in[1];
    const float* W1   = (const float*)d_in[2];
    const float* b1   = (const float*)d_in[3];
    const float* W2   = (const float*)d_in[4];
    const float* b2   = (const float*)d_in[5];
    const float* Wr1  = (const float*)d_in[6];
    const float* br1  = (const float*)d_in[7];
    const float* Wr2  = (const float*)d_in[8];
    const float* br2  = (const float*)d_in[9];
    const float* rs1  = (const float*)d_in[10];
    const float* rs2  = (const float*)d_in[11];
    float* out = (float*)d_out;

    const int threads = 256;                 // 8 waves per block (wave32)
    const int wpb     = threads / 32;
    int blocks = 2048;                       // 16384 waves -> 16 samples/wave
    if (blocks * wpb > B) blocks = (B + wpb - 1) / wpb;
    const int nWaves = blocks * wpb;

    knn_logit_head_kernel<<<blocks, threads, 0, stream>>>(
        vals, clsf, W1, b1, W2, b2, Wr1, br1, Wr2, br2, rs1, rs2,
        out, B, nWaves);
}
